// SelfAtt_29085518529014
// MI455X (gfx1250) — matrix operand order repaired and checked by reading, not compiled, not run on hardware
//
#include <hip/hip_runtime.h>

typedef __attribute__((ext_vector_type(16))) __bf16 v16bf;
typedef __attribute__((ext_vector_type(8)))  __bf16 v8bf;
typedef __attribute__((ext_vector_type(2)))  __bf16 v2bf;
typedef __attribute__((ext_vector_type(8)))  float  v8f;
typedef int v4i32 __attribute__((vector_size(4 * sizeof(int))));

#define EMB   768
#define HEADS 12
#define HDIM  64
#define SEQ   2048
#define BATCH 2
#define ROWS  (BATCH * SEQ)   // 4096
#define QKVN  (3 * EMB)       // 2304

// Detect the gfx1250 async global->LDS builtins (ASYNCcnt path).
#if defined(__has_builtin)
#  if __has_builtin(__builtin_amdgcn_global_load_async_to_lds_b128) && \
      __has_builtin(__builtin_amdgcn_s_wait_asynccnt)
#    define USE_ASYNC_LDS 1
#  endif
#endif
#ifndef USE_ASYNC_LDS
#  define USE_ASYNC_LDS 0
#endif

#if USE_ASYNC_LDS
static __device__ __forceinline__ void async_cp_b128(const __bf16* gsrc, __bf16* ldst) {
  __builtin_amdgcn_global_load_async_to_lds_b128(
      (__attribute__((address_space(1))) v4i32*)gsrc,
      (__attribute__((address_space(3))) v4i32*)ldst,
      0, 0);
}
#endif

// ---------------------------------------------------------------------------
// WMMA fragment helpers (contiguous k order: lane half h holds k = 16h..16h+15 for A and B)
// ---------------------------------------------------------------------------

// A-matrix 16x32 bf16: lanes 0-15 row=lane, K0..15
//                      lanes 16-31 row=lane-16, K16..31
static __device__ __forceinline__ v16bf load_a16x32(const __bf16* base, int stride, int lane) {
  int r = lane & 15, hi = lane >> 4;
  const __bf16* p = base + r * stride + hi * 16;
  v8bf lo = *(const v8bf*)p;          // K = hi*16 + 0..7
  v8bf hh = *(const v8bf*)(p + 8);    // K = hi*16 + 8..15
  v16bf o;
#pragma unroll
  for (int i = 0; i < 8; ++i) { o[i] = lo[i]; o[i + 8] = hh[i]; }
  return o;
}

// B-matrix 32x16 bf16 stored as rows-of-N (each "row" n holds K contiguous):
// lanes 0-15: col=lane, K0..15 ; lanes 16-31: col=lane-16, K16..31
static __device__ __forceinline__ v16bf load_b32x16(const __bf16* base, int stride, int lane) {
  int n = lane & 15, hi = lane >> 4;
  return *(const v16bf*)(base + n * stride + hi * 16);
}

static __device__ __forceinline__ v8f wmma_bf16(v16bf a, v16bf b, v8f c) {
  return __builtin_amdgcn_wmma_f32_16x16x32_bf16(false, a, false, b, (short)0, c,
                                                 false, false);
}

// ---------------------------------------------------------------------------
// LayerNorm (+f32 -> bf16): one wave32 per 768-wide row, 8 rows per block
// ---------------------------------------------------------------------------
__global__ __launch_bounds__(256) void ln_kernel(const float* __restrict__ x,
                                                 const float* __restrict__ g,
                                                 const float* __restrict__ bta,
                                                 __bf16* __restrict__ out) {
  int lane = threadIdx.x & 31;
  int row  = blockIdx.x * 8 + (threadIdx.x >> 5);
  const float* xr = x + (size_t)row * EMB;
  float v[24];
  float s = 0.f;
#pragma unroll
  for (int i = 0; i < 24; ++i) { v[i] = xr[lane + i * 32]; s += v[i]; }
#pragma unroll
  for (int m = 16; m >= 1; m >>= 1) s += __shfl_xor(s, m, 32);
  float mu = s * (1.0f / EMB);
  float var = 0.f;
#pragma unroll
  for (int i = 0; i < 24; ++i) { float d = v[i] - mu; var += d * d; }
#pragma unroll
  for (int m = 16; m >= 1; m >>= 1) var += __shfl_xor(var, m, 32);
  float rs = rsqrtf(var * (1.0f / EMB) + 1e-5f);
  __bf16* orow = out + (size_t)row * EMB;
#pragma unroll
  for (int i = 0; i < 24; ++i) {
    int c = lane + i * 32;
    orow[c] = (__bf16)((v[i] - mu) * rs * g[c] + bta[c]);
  }
}

// ---------------------------------------------------------------------------
// f32 -> bf16 conversion (weights)
// ---------------------------------------------------------------------------
__global__ void cvt_kernel(const float* __restrict__ in, __bf16* __restrict__ out, int n) {
  int stride = gridDim.x * blockDim.x;
  for (int i = blockIdx.x * blockDim.x + threadIdx.x; i < n; i += stride)
    out[i] = (__bf16)in[i];
}

// ---------------------------------------------------------------------------
// Tile staging for the GEMMs.  Tiles are 128x32 bf16 (8KB) with padded LDS
// stride; 256 threads stage a tile as one 32B slice per thread.
// Async path: 2 x GLOBAL_LOAD_ASYNC_TO_LDS_B128 per tile per thread.
// ---------------------------------------------------------------------------
#define LDA 40
#define LDB 40

#if USE_ASYNC_LDS
#define STAGE_TILES(Abuf, Bbuf, k0)                                            \
  do {                                                                         \
    const __bf16* ga = A + (size_t)(bm + lrow) * EMB + (k0) + lcol;            \
    const __bf16* gb = W + (size_t)(bn + lrow) * EMB + (k0) + lcol;            \
    __bf16* la = (Abuf) + lrow * LDA + lcol;                                   \
    __bf16* lb = (Bbuf) + lrow * LDB + lcol;                                   \
    async_cp_b128(ga, la);                                                     \
    async_cp_b128(ga + 8, la + 8);                                             \
    async_cp_b128(gb, lb);                                                     \
    async_cp_b128(gb + 8, lb + 8);                                             \
  } while (0)
#else
#define STAGE_TILES(Abuf, Bbuf, k0)                                            \
  do {                                                                         \
    v16bf ta = *(const v16bf*)(A + (size_t)(bm + lrow) * EMB + (k0) + lcol);   \
    v16bf tb = *(const v16bf*)(W + (size_t)(bn + lrow) * EMB + (k0) + lcol);   \
    *(v16bf*)((Abuf) + lrow * LDA + lcol) = ta;                                \
    *(v16bf*)((Bbuf) + lrow * LDB + lcol) = tb;                                \
  } while (0)
#endif

// Waits for the *current* tile: with a next tile in flight there are 8
// outstanding async ops, the first (in-order) 4 are the current tile.
#if USE_ASYNC_LDS
#define WAIT_STAGE(more)                                                       \
  do {                                                                         \
    if (more) __builtin_amdgcn_s_wait_asynccnt(4);                             \
    else      __builtin_amdgcn_s_wait_asynccnt(0);                             \
  } while (0)
#else
#define WAIT_STAGE(more) do { } while (0)
#endif

// ---------------------------------------------------------------------------
// QKV GEMM: [4096,768]bf16 x [2304,768]bf16^T (+bias), scatter to q/k/v.
// Block tile 128x128x32, 256 threads = 8 waves in 4(M) x 2(N); wave = 32x64.
// LDS double-buffered; next tile streams in via ASYNCcnt while WMMAs run.
// qkv column j factors as j = h*192 + d*3 + which   (einops 'b n (h d qkv)')
// ---------------------------------------------------------------------------
__global__ __launch_bounds__(256) void qkv_gemm_kernel(
    const __bf16* __restrict__ A, const __bf16* __restrict__ W,
    const float* __restrict__ bias,
    __bf16* __restrict__ qo, __bf16* __restrict__ ko, __bf16* __restrict__ vo) {
  __shared__ __bf16 As[2][128 * LDA];
  __shared__ __bf16 Bs[2][128 * LDB];
  int tid = threadIdx.x;
  int lane = tid & 31, wave = tid >> 5;
  int waveM = wave & 3, waveN = wave >> 2;
  int bm = blockIdx.x * 128, bn = blockIdx.y * 128;
  int lrow = tid >> 1, lcol = (tid & 1) * 16;
  v8f acc[2][4] = {};

  STAGE_TILES(As[0], Bs[0], 0);
  int buf = 0;
  for (int k0 = 0; k0 < EMB; k0 += 32, buf ^= 1) {
    bool more = (k0 + 32 < EMB);
    if (more) STAGE_TILES(As[buf ^ 1], Bs[buf ^ 1], k0 + 32);
    WAIT_STAGE(more);
    __syncthreads();
    v16bf af[2], bf[4];
#pragma unroll
    for (int mf = 0; mf < 2; ++mf)
      af[mf] = load_a16x32(As[buf] + (waveM * 32 + mf * 16) * LDA, LDA, lane);
#pragma unroll
    for (int nf = 0; nf < 4; ++nf)
      bf[nf] = load_b32x16(Bs[buf] + (waveN * 64 + nf * 16) * LDB, LDB, lane);
#pragma unroll
    for (int mf = 0; mf < 2; ++mf)
#pragma unroll
      for (int nf = 0; nf < 4; ++nf)
        acc[mf][nf] = wmma_bf16(af[mf], bf[nf], acc[mf][nf]);
    __syncthreads();
  }
  int hi = lane >> 4, ln = lane & 15;
#pragma unroll
  for (int mf = 0; mf < 2; ++mf)
#pragma unroll
    for (int nf = 0; nf < 4; ++nf)
#pragma unroll
      for (int e = 0; e < 8; ++e) {
        int i = bm + waveM * 32 + mf * 16 + e + hi * 8;
        int j = bn + waveN * 64 + nf * 16 + ln;
        float val = acc[mf][nf][e] + bias[j];
        int which = j % 3, hd = j / 3;
        int dd = hd & 63, hh = hd >> 6;
        int b = i >> 11, n = i & 2047;
        size_t dst = (((size_t)(b * HEADS + hh)) * SEQ + n) * HDIM + dd;
        __bf16 bv = (__bf16)val;
        if (which == 0)      qo[dst] = bv;
        else if (which == 1) ko[dst] = bv;
        else                 vo[dst] = bv;
      }
}

// ---------------------------------------------------------------------------
// Flash attention: grid (32 qblocks, 24 b*h). 128 thr = 4 waves x 16 q-rows.
// scores = (Q K^T) * sqrt(d)  (reference divides by d^-0.5), online softmax.
// ---------------------------------------------------------------------------
__global__ __launch_bounds__(128) void attn_kernel(
    const __bf16* __restrict__ qm, const __bf16* __restrict__ km,
    const __bf16* __restrict__ vm, __bf16* __restrict__ ao) {
  __shared__ __bf16 Vt[64 * 72];      // transposed V tile: Vt[d][key]
  __shared__ __bf16 Pl[4][16 * 72];   // per-wave probabilities
  int tid = threadIdx.x;
  int lane = tid & 31, wave = tid >> 5;
  int hi = lane >> 4, ln = lane & 15;
  int bh = blockIdx.y;
  int qrow0 = blockIdx.x * 64 + wave * 16;
  const size_t base = (size_t)bh * SEQ * HDIM;

  v16bf qf[2];
#pragma unroll
  for (int kt = 0; kt < 2; ++kt) {
    const __bf16* p = qm + base + (size_t)(qrow0 + ln) * HDIM + kt * 32 + hi * 16;
    v8bf lo = *(const v8bf*)p;
    v8bf hh = *(const v8bf*)(p + 8);
#pragma unroll
    for (int i = 0; i < 8; ++i) { qf[kt][i] = lo[i]; qf[kt][i + 8] = hh[i]; }
  }

  float mi[8], li[8];
  v8f o[4] = {};
#pragma unroll
  for (int e = 0; e < 8; ++e) { mi[e] = -1e30f; li[e] = 0.f; }

  for (int kb = 0; kb < SEQ; kb += 64) {
    __syncthreads();
#pragma unroll
    for (int it = 0; it < 16; ++it) {            // cooperative V transpose
      int e = it * 128 + tid;                    // 2 d-elems per thread-step
      int key = e >> 5, d2 = (e & 31) * 2;
      v2bf t = *(const v2bf*)(vm + base + (size_t)(kb + key) * HDIM + d2);
      Vt[d2 * 72 + key]       = t[0];
      Vt[(d2 + 1) * 72 + key] = t[1];
    }
    __syncthreads();

    v8f s[4];
#pragma unroll
    for (int nf = 0; nf < 4; ++nf) {
      v8f c = {};
#pragma unroll
      for (int kt = 0; kt < 2; ++kt) {
        v16bf kf = *(const v16bf*)(km + base +
                     (size_t)(kb + nf * 16 + ln) * HDIM + kt * 32 + hi * 16);
        c = wmma_bf16(qf[kt], kf, c);
      }
      s[nf] = c;
    }

    float mnew[8];
#pragma unroll
    for (int e = 0; e < 8; ++e) mnew[e] = mi[e];
#pragma unroll
    for (int nf = 0; nf < 4; ++nf)
#pragma unroll
      for (int e = 0; e < 8; ++e) {
        float sv = s[nf][e] * 8.0f;              // * sqrt(64)
        s[nf][e] = sv;
        mnew[e] = fmaxf(mnew[e], sv);
      }
#pragma unroll
    for (int msk = 1; msk < 16; msk <<= 1)
#pragma unroll
      for (int e = 0; e < 8; ++e)
        mnew[e] = fmaxf(mnew[e], __shfl_xor(mnew[e], msk, 32));

    float alpha[8], rsum[8];
#pragma unroll
    for (int e = 0; e < 8; ++e) { alpha[e] = __expf(mi[e] - mnew[e]); mi[e] = mnew[e]; rsum[e] = 0.f; }
#pragma unroll
    for (int nf = 0; nf < 4; ++nf)
#pragma unroll
      for (int e = 0; e < 8; ++e) {
        float p = __expf(s[nf][e] - mnew[e]);
        s[nf][e] = p;
        rsum[e] += p;
      }
#pragma unroll
    for (int msk = 1; msk < 16; msk <<= 1)
#pragma unroll
      for (int e = 0; e < 8; ++e)
        rsum[e] += __shfl_xor(rsum[e], msk, 32);
#pragma unroll
    for (int e = 0; e < 8; ++e) li[e] = li[e] * alpha[e] + rsum[e];
#pragma unroll
    for (int df = 0; df < 4; ++df)
#pragma unroll
      for (int e = 0; e < 8; ++e) o[df][e] *= alpha[e];

    __bf16* pl = Pl[wave];                       // C-frag -> A-frag via LDS
#pragma unroll
    for (int nf = 0; nf < 4; ++nf)
#pragma unroll
      for (int e = 0; e < 8; ++e)
        pl[(e + hi * 8) * 72 + nf * 16 + ln] = (__bf16)s[nf][e];

    v16bf pf[2];
#pragma unroll
    for (int kt = 0; kt < 2; ++kt) {
      const __bf16* p = pl + ln * 72 + kt * 32 + hi * 16;
      v8bf lo = *(const v8bf*)p;
      v8bf hh = *(const v8bf*)(p + 8);
#pragma unroll
      for (int i = 0; i < 8; ++i) { pf[kt][i] = lo[i]; pf[kt][i + 8] = hh[i]; }
    }
#pragma unroll
    for (int df = 0; df < 4; ++df)
#pragma unroll
      for (int kt = 0; kt < 2; ++kt) {
        v16bf vf = *(const v16bf*)(Vt + (df * 16 + ln) * 72 + kt * 32 + hi * 16);
        o[df] = wmma_bf16(pf[kt], vf, o[df]);
      }
  }

  int b = bh / HEADS, h = bh % HEADS;
#pragma unroll
  for (int df = 0; df < 4; ++df)
#pragma unroll
    for (int e = 0; e < 8; ++e) {
      float val = o[df][e] / li[e];
      int n = qrow0 + e + hi * 8;
      int col = h * HDIM + df * 16 + ln;
      ao[((size_t)(b * SEQ + n)) * EMB + col] = (__bf16)val;
    }
}

// ---------------------------------------------------------------------------
// Output projection: y = x + attOut @ Wp^T + b  (f32 out)
// ---------------------------------------------------------------------------
__global__ __launch_bounds__(256) void proj_gemm_kernel(
    const __bf16* __restrict__ A, const __bf16* __restrict__ W,
    const float* __restrict__ bias, const float* __restrict__ resid,
    float* __restrict__ out) {
  __shared__ __bf16 As[2][128 * LDA];
  __shared__ __bf16 Bs[2][128 * LDB];
  int tid = threadIdx.x;
  int lane = tid & 31, wave = tid >> 5;
  int waveM = wave & 3, waveN = wave >> 2;
  int bm = blockIdx.x * 128, bn = blockIdx.y * 128;
  int lrow = tid >> 1, lcol = (tid & 1) * 16;
  v8f acc[2][4] = {};

  STAGE_TILES(As[0], Bs[0], 0);
  int buf = 0;
  for (int k0 = 0; k0 < EMB; k0 += 32, buf ^= 1) {
    bool more = (k0 + 32 < EMB);
    if (more) STAGE_TILES(As[buf ^ 1], Bs[buf ^ 1], k0 + 32);
    WAIT_STAGE(more);
    __syncthreads();
    v16bf af[2], bf[4];
#pragma unroll
    for (int mf = 0; mf < 2; ++mf)
      af[mf] = load_a16x32(As[buf] + (waveM * 32 + mf * 16) * LDA, LDA, lane);
#pragma unroll
    for (int nf = 0; nf < 4; ++nf)
      bf[nf] = load_b32x16(Bs[buf] + (waveN * 64 + nf * 16) * LDB, LDB, lane);
#pragma unroll
    for (int mf = 0; mf < 2; ++mf)
#pragma unroll
      for (int nf = 0; nf < 4; ++nf)
        acc[mf][nf] = wmma_bf16(af[mf], bf[nf], acc[mf][nf]);
    __syncthreads();
  }
  int hi = lane >> 4, ln = lane & 15;
#pragma unroll
  for (int mf = 0; mf < 2; ++mf)
#pragma unroll
    for (int nf = 0; nf < 4; ++nf)
#pragma unroll
      for (int e = 0; e < 8; ++e) {
        int i = bm + waveM * 32 + mf * 16 + e + hi * 8;
        int j = bn + waveN * 64 + nf * 16 + ln;
        size_t idx = (size_t)i * EMB + j;
        out[idx] = resid[idx] + acc[mf][nf][e] + bias[j];
      }
}

// ---------------------------------------------------------------------------
// Host launcher
// ---------------------------------------------------------------------------
extern "C" void kernel_launch(void* const* d_in, const int* in_sizes, int n_in,
                              void* d_out, int out_size, void* d_ws, size_t ws_size,
                              hipStream_t stream) {
  (void)in_sizes; (void)n_in; (void)out_size; (void)ws_size;
  const size_t XE = (size_t)ROWS * EMB;        // 4096*768 elems
  char* ws = (char*)d_ws;
  auto bump = [&](size_t bytes) -> char* {
    char* p = ws;
    ws += (bytes + 255) & ~(size_t)255;
    return p;
  };

  float* out = (float*)d_out;

  for (int m = 0; m < 2; ++m) {
    const float* x      = (const float*)d_in[m];
    const float* ln_g   = (const float*)d_in[2 + m * 6 + 0];
    const float* ln_b   = (const float*)d_in[2 + m * 6 + 1];
    const float* w_qkv  = (const float*)d_in[2 + m * 6 + 2];
    const float* b_qkv  = (const float*)d_in[2 + m * 6 + 3];
    const float* w_proj = (const float*)d_in[2 + m * 6 + 4];
    const float* b_proj = (const float*)d_in[2 + m * 6 + 5];

    __bf16* xln    = (__bf16*)bump(XE * 2);
    __bf16* wqkvb  = (__bf16*)bump((size_t)QKVN * EMB * 2);
    __bf16* wprojb = (__bf16*)bump((size_t)EMB * EMB * 2);
    __bf16* qb     = (__bf16*)bump(XE * 2);
    __bf16* kb     = (__bf16*)bump(XE * 2);
    __bf16* vb     = (__bf16*)bump(XE * 2);
    __bf16* attout = (__bf16*)bump(XE * 2);

    ln_kernel<<<ROWS / 8, 256, 0, stream>>>(x, ln_g, ln_b, xln);
    cvt_kernel<<<864, 256, 0, stream>>>(w_qkv, wqkvb, QKVN * EMB);
    cvt_kernel<<<288, 256, 0, stream>>>(w_proj, wprojb, EMB * EMB);
    qkv_gemm_kernel<<<dim3(ROWS / 128, QKVN / 128), 256, 0, stream>>>(
        xln, wqkvb, b_qkv, qb, kb, vb);
    attn_kernel<<<dim3(SEQ / 64, BATCH * HEADS), 128, 0, stream>>>(
        qb, kb, vb, attout);
    proj_gemm_kernel<<<dim3(ROWS / 128, EMB / 128), 256, 0, stream>>>(
        attout, wprojb, b_proj, x, out + (size_t)m * XE);
  }
}
